// CholecMetric_26998164422908
// MI455X (gfx1250) — 8-bit matrix operand order repaired and checked by reading, compiled, not run on hardware
//
#include <hip/hip_runtime.h>
#include <hip/hip_bf16.h>

#define NB   8
#define NP   32
#define NT   16
#define HW   65536
#define PKW  (HW / 32)
#define KCHUNK 1024
#define NCHUNK (HW / KCHUNK)

typedef __attribute__((ext_vector_type(8))) int v8i;

__device__ __forceinline__ unsigned pack01(int4 w) {
    return (unsigned)(w.x != 0)        |
           ((unsigned)(w.y != 0) << 8) |
           ((unsigned)(w.z != 0) << 16)|
           ((unsigned)(w.w != 0) << 24);
}

__device__ __forceinline__ unsigned nib4(unsigned t) {
    return (t | (t >> 7) | (t >> 14) | (t >> 21)) & 0xFu;
}

__global__ void cm_zero_kernel(int* __restrict__ p) {
    int i = blockIdx.x * 256 + threadIdx.x;
    if (i < NB * NP * NT + NB * NT) p[i] = 0;
}

template <int DO_PACK>
__global__ void cm_inters_wmma_kernel(const int* __restrict__ Gp,
                                      const int* __restrict__ Gt,
                                      int* __restrict__ inters,
                                      int* __restrict__ area,
                                      unsigned* __restrict__ packed) {
    const int lane = threadIdx.x;
    const int half = lane >> 4;
    const int low  = lane & 15;
    const int chunk = blockIdx.x;
    const int n     = blockIdx.y;

    const int* a0Row = Gp + (size_t)(n * NP + low) * HW;
    const int* a1Row = Gp + (size_t)(n * NP + 16 + low) * HW;
    const int* bRow  = Gt + (size_t)(n * NT + low) * HW;
    const int  kstart = chunk * KCHUNK;

    int aK[8], bK[8];
#pragma unroll
    for (int j = 0; j < 8; ++j) {
        aK[j] = (j >> 1) * 16 + (j & 1) * 4 + half * 8;
        bK[j] = (j >> 1) * 16 + (j & 1) * 4 + half * 8;
    }

    unsigned* pk0 = packed + (size_t)(n * NP + low) * PKW;
    unsigned* pk1 = packed + (size_t)(n * NP + 16 + low) * PKW;

    v8i acc0 = {}, acc1 = {};
    int areaCnt = 0;
    for (int kk = 0; kk < KCHUNK / 64; ++kk) {
        const int kbase = kstart + (kk << 6);
        v8i a0, a1, b;
#pragma unroll
        for (int j = 0; j < 8; ++j) {
            a0[j] = (int)pack01(*(const int4*)(a0Row + kbase + aK[j]));
            a1[j] = (int)pack01(*(const int4*)(a1Row + kbase + aK[j]));
            b[j]  = (int)pack01(*(const int4*)(bRow  + kbase + bK[j]));
        }

        if (DO_PACK) {
            unsigned pw0 = 0, pw1 = 0;
#pragma unroll
            for (int j = 0; j < 8; ++j) {
                pw0 |= nib4((unsigned)a0[j]) << (j * 4);
                pw1 |= nib4((unsigned)a1[j]) << (j * 4);
            }
            const int widx = (kbase >> 6) * 2 + half;
            pk0[widx] = pw0;
            pk1[widx] = pw1;
        }

#pragma unroll
        for (int j = 0; j < 8; ++j) areaCnt += __popc((unsigned)b[j]);

        acc0 = __builtin_amdgcn_wmma_i32_16x16x64_iu8(false, a0, false, b, acc0, false, false);
        acc1 = __builtin_amdgcn_wmma_i32_16x16x64_iu8(false, a1, false, b, acc1, false, false);
    }

    int* dst0 = inters + ((n * NP) * NT) + low;
    int* dst1 = inters + ((n * NP + 16) * NT) + low;
#pragma unroll
    for (int v = 0; v < 8; ++v) {
        atomicAdd(dst0 + (v + 8 * half) * NT, acc0[v]);
        atomicAdd(dst1 + (v + 8 * half) * NT, acc1[v]);
    }
    atomicAdd(area + n * NT + low, areaCnt);
}

__global__ void cm_iogmax_kernel(const int* __restrict__ inters,
                                 const int* __restrict__ area,
                                 float* __restrict__ iogm) {
    const int idx = threadIdx.x;
    const int n = idx / NP;
    float best = 0.0f;
#pragma unroll
    for (int t = 0; t < NT; ++t) {
        int ga = area[n * NT + t];
        if (ga > 0) {
            float v = (float)inters[idx * NT + t] / (float)ga;
            best = fmaxf(best, v);
        }
    }
    iogm[idx] = best;
}

__global__ void cm_score_packed_kernel(const unsigned* __restrict__ packed,
                                       const float* __restrict__ iogm,
                                       float* __restrict__ out) {
    const int idx = blockIdx.x * 256 + threadIdx.x;
    const int n   = idx >> 16;
    const int pix = idx & (HW - 1);
    const int o = pix & 63;
    const int pixword = (pix >> 6) * 2 + ((pix >> 3) & 1);
    const int bit = (((o >> 4) * 2 + ((o >> 2) & 1)) << 2) + (o & 3);
    const unsigned* base = packed + (size_t)n * NP * PKW + pixword;
    const float* im = iogm + n * NP;
    float s = 0.0f;
    int cover = 0;
#pragma unroll
    for (int p = 0; p < NP; ++p) {
        if ((base[(size_t)p * PKW] >> bit) & 1u) { s += im[p]; ++cover; }
    }
    out[idx] = (cover > 0) ? (s / (float)cover) : 0.0f;
}

__global__ void cm_score_kernel(const int* __restrict__ Gp,
                                const float* __restrict__ iogm,
                                float* __restrict__ out) {
    const int idx = blockIdx.x * 256 + threadIdx.x;
    const int n   = idx >> 16;
    const int pix = idx & (HW - 1);
    const int* gp = Gp + (size_t)n * NP * HW + pix;
    const float* im = iogm + n * NP;
    float s = 0.0f;
    int cover = 0;
#pragma unroll
    for (int p = 0; p < NP; ++p) {
        if (gp[(size_t)p * HW] != 0) { s += im[p]; ++cover; }
    }
    out[idx] = (cover > 0) ? (s / (float)cover) : 0.0f;
}

extern "C" void kernel_launch(void* const* d_in, const int* in_sizes, int n_in,
                              void* d_out, int out_size, void* d_ws, size_t ws_size,
                              hipStream_t stream) {
    const int* Gp = (const int*)d_in[0];
    const int* Gt = (const int*)d_in[1];
    float* out = (float*)d_out;

    int*      inters = (int*)d_ws;
    int*      area   = inters + NB * NP * NT;
    float*    iogm   = (float*)(area + NB * NT);
    unsigned* packed = (unsigned*)(iogm + NB * NP);

    const size_t need_packed =
        (size_t)(NB * NP * NT + NB * NT + NB * NP) * 4 +
        (size_t)NB * NP * PKW * 4;
    const int do_pack = (ws_size >= need_packed) ? 1 : 0;

    (void)in_sizes; (void)n_in; (void)out_size;

    cm_zero_kernel<<<dim3(17), dim3(256), 0, stream>>>(inters);

    if (do_pack) {
        cm_inters_wmma_kernel<1><<<dim3(NCHUNK, NB), dim3(32), 0, stream>>>(
            Gp, Gt, inters, area, packed);
    } else {
        cm_inters_wmma_kernel<0><<<dim3(NCHUNK, NB), dim3(32), 0, stream>>>(
            Gp, Gt, inters, area, packed);
    }

    cm_iogmax_kernel<<<dim3(1), dim3(256), 0, stream>>>(inters, area, iogm);

    if (do_pack) {
        cm_score_packed_kernel<<<dim3(NB * HW / 256), dim3(256), 0, stream>>>(
            packed, iogm, out);
    } else {
        cm_score_kernel<<<dim3(NB * HW / 256), dim3(256), 0, stream>>>(
            Gp, iogm, out);
    }
}
